// SNNtorch_ConvFCLIF_11132555231297
// MI455X (gfx1250) — matrix operand order repaired and checked by reading, not compiled, not run on hardware
//
#include <hip/hip_runtime.h>
#include <hip/hip_bf16.h>
#include <stdint.h>

// Fused SNN step for MI455X (gfx1250, wave32, WMMA).
//
// Roofline: ~270 MB mandatory HBM traffic (~11.6 us @ 23.3 TB/s) vs ~9.3 GFLOP
// => memory-bound if the two 128x128 FC layers run on the matrix pipe.
// Single fused kernel: spikes/activations live entirely in LDS.
// FC layers: v_wmma_f32_16x16x32_f16 (spikes {0,1} exact in f16, f32 accum).
// mem0 == 0 in the reference, so beta is dead: mem=x; spk=(x>thr); mem*=(1-spk).
//
//  * LDS row padding: 272B f16 / 528B f32 row strides (4-bank skew) to kill
//    the 16-way same-bank conflicts a 256B stride would cause on every
//    ds_load_b128 feeding the WMMAs.
//  * LDS 136.5KB total -> 2 blocks/WGP (vs 1 at 166KB): one block's HBM
//    staging overlaps the other's WMMA phase.
//  * CDNA5 async staging: global_load_async_to_lds_b128 (ASYNCcnt-tracked,
//    no VGPR round-trip). Signature from hipcc diagnostic: (int4 AS1*,
//    int4 AS3*, imm offset, imm cpol). Guarded by __has_builtin.

typedef __attribute__((ext_vector_type(16))) _Float16 v16h;
typedef __attribute__((ext_vector_type(8)))  _Float16 v8h;
typedef __attribute__((ext_vector_type(8)))  float    v8f;

#define THREADS 256
#define ROWS_PER_BLOCK 128   // 8 waves x 16 batch rows
#define F 128
#define SIN_STR 132          // f32 input row stride (528B == 4-bank skew)
#define HSTR 136             // f16 row stride (272B == 4-bank skew)

// LDS carve (dynamic shared), all offsets 512B aligned:
//   s_in   : 128 * 132 f32 = 67584 B   (input staging; reused as s_w2 later)
//   s_w1   : 128 * 136 f16 = 34816 B
//   s_spk  : 128 * 136 f16 = 34816 B
//   s_w3   : 2*128 f32     =  1024 B
//   thr1/thr2/thrc         :   512 B each
#define OFF_W1   67584
#define OFF_SPK  102400
#define OFF_W3   137216
#define OFF_THR1 138240
#define OFF_THR2 138752
#define OFF_THRC 139264
#define SMEM_BYTES 139776

#if defined(__has_builtin)
#if __has_builtin(__builtin_amdgcn_global_load_async_to_lds_b128)
#define HAVE_ASYNC_LDS 1
#endif
#endif

#if HAVE_ASYNC_LDS
typedef int v4i __attribute__((vector_size(16)));
typedef __attribute__((address_space(1))) v4i GV4;   // global int4
typedef __attribute__((address_space(3))) v4i LV4;   // LDS int4
#endif

__global__ void __launch_bounds__(THREADS)
snn_fused_kernel(const float* __restrict__ in,       // B*2*8*8
                 const float* __restrict__ conv_w,   // 2*2*3*3
                 const float* __restrict__ conv_b,   // 2
                 const float* __restrict__ thr_conv, // 128
                 const float* __restrict__ fc1_w,    // 128*128 (o-major)
                 const float* __restrict__ thr1,     // 128
                 const float* __restrict__ fc2_w,    // 128*128
                 const float* __restrict__ thr2,     // 128
                 const float* __restrict__ fc3_w,    // 2*128
                 float* __restrict__ out,            // B*2
                 float* __restrict__ mem_c,          // B*128
                 float* __restrict__ mem1,           // B*128
                 float* __restrict__ mem2)           // B*128
{
    extern __shared__ char smem[];
    float*    s_in   = (float*)smem;                      // phase 1 only
    _Float16* s_w2   = (_Float16*)smem;                   // alias, phases 2+
    _Float16* s_w1   = (_Float16*)(smem + OFF_W1);
    _Float16* s_spk  = (_Float16*)(smem + OFF_SPK);
    float*    s_w3   = (float*)(smem + OFF_W3);
    float*    s_thr1 = (float*)(smem + OFF_THR1);
    float*    s_thr2 = (float*)(smem + OFF_THR2);
    float*    s_thrc = (float*)(smem + OFF_THRC);

    const int t    = threadIdx.x;
    const int row0 = blockIdx.x * ROWS_PER_BLOCK;   // first batch row of block

    // ---------------- Phase 0: stage input + fc1 weights -------------------
    // Input: 128 rows x 32 float4, padded LDS rows (r*33 float4).
#if HAVE_ASYNC_LDS
    {
        const float4* gin = (const float4*)(in + (size_t)row0 * F);
        for (int i = t; i < ROWS_PER_BLOCK * (F / 4); i += THREADS) {
            const int r = i >> 5, c = i & 31;
            // AS3 ptr value is the 32-bit LDS offset (generic addr low bits).
            __builtin_amdgcn_global_load_async_to_lds_b128(
                (GV4*)(uintptr_t)(gin + i),
                (LV4*)(uint32_t)(uintptr_t)(s_in + r * SIN_STR + c * 4),
                0, 0);
        }
    }
#else
    {
        const float4* gin = (const float4*)(in + (size_t)row0 * F);
        float4* sin4 = (float4*)s_in;
        for (int i = t; i < ROWS_PER_BLOCK * (F / 4); i += THREADS) {
            const int r = i >> 5, c = i & 31;
            sin4[r * 33 + c] = gin[i];
        }
    }
#endif
    for (int i = t; i < F * F; i += THREADS) {
        const int r = i >> 7, c = i & 127;
        s_w1[r * HSTR + c] = (_Float16)fc1_w[i];    // [o][k] row-major
    }
    for (int i = t; i < 2 * F; i += THREADS) s_w3[i] = fc3_w[i];
    if (t < F) {
        s_thr1[t] = thr1[t];
        s_thr2[t] = thr2[t];
        s_thrc[t] = thr_conv[t];
    }
#if HAVE_ASYNC_LDS
#if __has_builtin(__builtin_amdgcn_s_wait_asynccnt)
    __builtin_amdgcn_s_wait_asynccnt(0);
#else
    asm volatile("s_wait_asynccnt 0x0" ::: "memory");
#endif
#endif
    __syncthreads();

    // ---------------- Phase 1: conv(2->2, 3x3, pad 1) + LIF ----------------
    // thread t handles (row = t/2, c_out = t&1): 64 outputs, fully unrolled so
    // all pad bounds-checks constant-fold away (968 FMAs/thread, LDS loads CSEd).
    {
        const int row = t >> 1;
        const int co  = t & 1;
        const float* xin = s_in + row * SIN_STR;    // [ci][y][x] = ci*64+y*8+x
        float wloc[18];
#pragma unroll
        for (int i = 0; i < 18; ++i) wloc[i] = conv_w[co * 18 + i];
        const float bias = conv_b[co];
        float*    gmem = mem_c + (size_t)(row0 + row) * F + co * 64;
        _Float16* sspk = s_spk + row * HSTR + co * 64;
#pragma unroll
        for (int y = 0; y < 8; ++y) {
#pragma unroll
            for (int x = 0; x < 8; ++x) {
                float acc = bias;
#pragma unroll
                for (int ci = 0; ci < 2; ++ci)
#pragma unroll
                    for (int ky = 0; ky < 3; ++ky) {
                        const int yy = y + ky - 1;
                        if (yy < 0 || yy > 7) continue;
#pragma unroll
                        for (int kx = 0; kx < 3; ++kx) {
                            const int xx = x + kx - 1;
                            if (xx < 0 || xx > 7) continue;
                            acc = fmaf(wloc[(ci * 3 + ky) * 3 + kx],
                                       xin[ci * 64 + yy * 8 + xx], acc);
                        }
                    }
                const int   f   = co * 64 + y * 8 + x;
                const float spk = (acc - s_thrc[f]) > 0.f ? 1.f : 0.f;
                gmem[y * 8 + x] = acc * (1.f - spk);   // mem_c (reset-to-zero)
                sspk[y * 8 + x] = (_Float16)spk;
            }
        }
    }
    __syncthreads();

    // -------- Phase 2a: convert fc2 weights into the (dead) input region ---
    // Overlaps GEMM1; published by the barrier between the two layers.
    for (int i = t; i < F * F; i += THREADS) {
        const int r = i >> 7, c = i & 127;
        s_w2[r * HSTR + c] = (_Float16)fc2_w[i];
    }

    // ---------------- Phase 2/3: two FC layers on WMMA ---------------------
    const int lane  = t & 31;
    const int wave  = t >> 5;
    const int r0    = wave * 16;          // wave-local batch rows [r0, r0+16)
    const int nlo   = lane & 15;
    const int half  = lane >> 4;          // 0: lanes 0-15, 1: lanes 16-31
    const int akoff = half * 8;           // A-tile K interleave (ISA 7.12.2)
    const int bkoff = half * 16;          // B-tile K half
    const int mbase = half * 8;           // C/D: rows M = mbase + vgpr

    auto gemm_lif = [&](const _Float16* sw, const float* sth,
                        float* __restrict__ gm) {
        v8f acc[8] = {};
#pragma unroll
        for (int kc = 0; kc < F; kc += 32) {
            // A-tile (16x32 f16): lane<16 holds K {kc..+7, kc+16..+23}
            union { v16h v; v8h h[2]; } a;
            const _Float16* pa = s_spk + (r0 + nlo) * HSTR + kc + akoff;
            a.h[0] = *(const v8h*)(pa);
            a.h[1] = *(const v8h*)(pa + 16);
#pragma unroll
            for (int n = 0; n < 8; ++n) {
                // B-tile (32x16 f16): N = n*16 + lane%16, K interleaved like A
                union { v16h v; v8h h[2]; } b;
                const _Float16* pb = sw + (n * 16 + nlo) * HSTR + kc + akoff;
                b.h[0] = *(const v8h*)(pb);
                b.h[1] = *(const v8h*)(pb + 16);
                acc[n] = __builtin_amdgcn_wmma_f32_16x16x32_f16(
                    false, a.v, false, b.v, (short)0, acc[n], false, false);
            }
        }
        // LIF on accumulators; spikes back to LDS (wave-local rows)
#pragma unroll
        for (int n = 0; n < 8; ++n) {
            const int   f   = n * 16 + nlo;
            const float thr = sth[f];
#pragma unroll
            for (int v = 0; v < 8; ++v) {
                const int   row = r0 + mbase + v;
                const float x   = acc[n][v];
                const float spk = (x - thr) > 0.f ? 1.f : 0.f;
                gm[(size_t)(row0 + row) * F + f] = x * (1.f - spk);
                s_spk[row * HSTR + f] = (_Float16)spk;
            }
        }
    };

    gemm_lif(s_w1, s_thr1, mem1);
    __syncthreads();                 // publish s_w2 (and spk, though wave-local)
    gemm_lif(s_w2, s_thr2, mem2);
    __syncthreads();

    // ---------------- Phase 4: readout 128 -> 2 ----------------------------
    {
        const int row = t >> 1;
        const int o   = t & 1;
        const v8h*   sp8 = (const v8h*)(s_spk + row * HSTR);
        const float* w3  = s_w3 + o * F;
        float acc = 0.f;
#pragma unroll
        for (int c = 0; c < 16; ++c) {
            const v8h s = sp8[c];
#pragma unroll
            for (int j = 0; j < 8; ++j)
                acc = fmaf((float)s[j], w3[c * 8 + j], acc);
        }
        out[(size_t)(row0 + row) * 2 + o] = acc;
    }
}

extern "C" void kernel_launch(void* const* d_in, const int* in_sizes, int n_in,
                              void* d_out, int out_size, void* d_ws, size_t ws_size,
                              hipStream_t stream) {
    (void)n_in; (void)out_size; (void)d_ws; (void)ws_size;
    // setup_inputs() order:
    // 0 input_, 1 conv_w, 2 conv_b, 3 beta_conv(dead), 4 thr_conv,
    // 5 fc1_w, 6 beta1(dead), 7 thr1, 8 fc2_w, 9 beta2(dead), 10 thr2, 11 fc3_w
    const float* in_      = (const float*)d_in[0];
    const float* conv_w   = (const float*)d_in[1];
    const float* conv_b   = (const float*)d_in[2];
    const float* thr_conv = (const float*)d_in[4];
    const float* fc1_w    = (const float*)d_in[5];
    const float* thr1     = (const float*)d_in[7];
    const float* fc2_w    = (const float*)d_in[8];
    const float* thr2     = (const float*)d_in[10];
    const float* fc3_w    = (const float*)d_in[11];

    const int B = in_sizes[0] / F;        // 131072

    float* out  = (float*)d_out;                     // (B,2,1,1) flat
    float* memc = out  + (size_t)B * 2;              // (B,128)
    float* m1   = memc + (size_t)B * F;              // (B,128)
    float* m2   = m1   + (size_t)B * F;              // (B,128)

    (void)hipFuncSetAttribute((const void*)snn_fused_kernel,
                              hipFuncAttributeMaxDynamicSharedMemorySize,
                              SMEM_BYTES);

    dim3 grid(B / ROWS_PER_BLOCK), block(THREADS);
    snn_fused_kernel<<<grid, block, SMEM_BYTES, stream>>>(
        in_, conv_w, conv_b, thr_conv, fc1_w, thr1, fc2_w, thr2, fc3_w,
        out, memc, m1, m2);
}
